// MultiHeadSelfAttention_63093069578879
// MI455X (gfx1250) — matrix operand order repaired and checked by reading, not compiled, not run on hardware
//
#include <hip/hip_runtime.h>

// MI455X (gfx1250) multi-head self-attention, bf16 WMMA + async global->LDS,
// register-blocked so each LDS B-fragment feeds 2-4 WMMAs.
// B=4, S=2048, D=1024, H=16, HD=64.
#define B_  4
#define S_  2048
#define D_  1024
#define H_  16
#define HD_ 64

typedef __attribute__((ext_vector_type(16))) __bf16 v16bf;
typedef __attribute__((ext_vector_type(2)))  __bf16 v2bf;
typedef __attribute__((ext_vector_type(8)))  float  v8f;

// round-to-nearest-even f32 -> bf16 (raw u16)
__device__ __forceinline__ unsigned short f2bf_u16(float f) {
  unsigned u = __builtin_bit_cast(unsigned, f);
  u += 0x7FFFu + ((u >> 16) & 1u);
  return (unsigned short)(u >> 16);
}

__device__ __forceinline__ void set_pair(v16bf& v, int i, unsigned u) {
  v2bf p = __builtin_bit_cast(v2bf, u);
  v[2 * i]     = p.x;
  v[2 * i + 1] = p.y;
}

// 16-bit A-matrix 16x32 layout (contiguous k order): dword i of lane l covers
// K = {k, k+1}, k = 2i + 16*(l/16). Always even.
__device__ __forceinline__ int a_k(int i, int hf) {
  return 2 * i + 16 * hf;
}

// ---- CDNA5 async global->LDS copy (ASYNCcnt path, ISA §15.18 op 98) -------
__device__ __forceinline__ void async_load_b128(void* lds_dst, const void* gsrc) {
  unsigned lds_off = (unsigned)(unsigned long long)lds_dst;  // low 32 = LDS offset
  unsigned long long ga = (unsigned long long)gsrc;
  asm volatile("global_load_async_to_lds_b128 %0, %1, off"
               :
               : "v"(lds_off), "v"(ga)
               : "memory");
}
__device__ __forceinline__ void wait_async() {
  asm volatile("s_wait_asynccnt 0" ::: "memory");
}

// ---------------------------------------------------------------------------
// Prep: x [B][S][D] f32 -> bf16 (so GEMM A-frags are pure b128 loads)
__global__ void prep_x_kernel(const float* __restrict__ x,
                              unsigned short* __restrict__ xb) {
  size_t i = (size_t)blockIdx.x * blockDim.x + threadIdx.x;
  if (i < (size_t)B_ * S_ * D_) xb[i] = f2bf_u16(x[i]);
}

// Prep: w_qkv [H][D][192] f32 -> w_qkv_t [H][192][D] bf16 (B-frag friendly)
__global__ void prep_wqkv_kernel(const float* __restrict__ wqkv,
                                 unsigned short* __restrict__ wt) {
  size_t idx = (size_t)blockIdx.x * blockDim.x + threadIdx.x;
  if (idx >= (size_t)H_ * 192 * D_) return;
  int h   = (int)(idx / (192 * D_));
  int rem = (int)(idx % (192 * D_));
  int e   = rem / D_;
  int d   = rem % D_;
  wt[idx] = f2bf_u16(wqkv[((size_t)h * D_ + d) * 192 + e]);
}

// Prep: w_out [D][D] f32 -> w_out_t [n][k] bf16
__global__ void prep_wout_kernel(const float* __restrict__ wout,
                                 unsigned short* __restrict__ wt) {
  size_t idx = (size_t)blockIdx.x * blockDim.x + threadIdx.x;
  if (idx >= (size_t)D_ * D_) return;
  int n = (int)(idx / D_);
  int k = (int)(idx % D_);
  wt[idx] = f2bf_u16(wout[(size_t)k * D_ + n]);
}

// ---------------------------------------------------------------------------
// QKV projection. Block = (b, h, 128-row s-block), 8 waves.
// Wave = 64 rows x 48 cols: 4 A-frags (global) x 3 B-frags (LDS), 12 WMMA per
// k-step; each LDS B-fragment is reused by 4 WMMAs.
__global__ __launch_bounds__(256) void qkv_proj_kernel(
    const unsigned short* __restrict__ Xb,  // [B][S][D] bf16
    const unsigned short* __restrict__ wt,  // [H][192][D] bf16
    unsigned short* __restrict__ Q,         // [B*H][S][HD] bf16
    unsigned short* __restrict__ K,         // [B*H][S][HD] bf16
    unsigned short* __restrict__ Vt)        // [B*H][HD][S] bf16 (transposed)
{
  __shared__ unsigned short wbuf[2][192 * 32];  // panel [192][32] bf16, 12KB x2

  const int tid = threadIdx.x;
  const int lane = tid & 31, wv = tid >> 5;
  const int hf = lane >> 4, l16 = lane & 15;
  const int rg = wv >> 2, cg = wv & 3;  // row-group (64 rows), col-group (48)
  const int sTiles = S_ / 128;
  const int bh = blockIdx.x / sTiles, st = blockIdx.x % sTiles;
  const int b = bh / H_, h = bh % H_;
  const int srow0 = st * 128 + rg * 64;

  const unsigned short* xbase = Xb + ((size_t)b * S_ + srow0) * D_;
  const unsigned short* wh = wt + (size_t)h * 192 * D_;

  auto stage_panel = [&](int k0, unsigned short* dst) {
#pragma unroll
    for (int j = 0; j < 3; ++j) {
      int c = tid + j * 256;  // 768 x 16B chunks
      async_load_b128(dst + (size_t)c * 8,
                      wh + (size_t)(c >> 2) * D_ + k0 + (c & 3) * 8);
    }
  };

  v8f acc[4][3] = {};
  stage_panel(0, wbuf[0]);
  wait_async();
  __syncthreads();
  int cur = 0;

  for (int k0 = 0; k0 < D_; k0 += 32) {
    if (k0 + 32 < D_) stage_panel(k0 + 32, wbuf[cur ^ 1]);

    v16bf a[4];
#pragma unroll
    for (int s = 0; s < 4; ++s) {
      const unsigned* apx =
          (const unsigned*)(xbase + (size_t)(s * 16 + l16) * D_ + k0);
#pragma unroll
      for (int i = 0; i < 8; ++i) set_pair(a[s], i, apx[a_k(i, hf) >> 1]);
    }
#pragma unroll
    for (int c = 0; c < 3; ++c) {
      int n = cg * 48 + c * 16 + l16;
      const unsigned* bp =
          (const unsigned*)&wbuf[cur][(size_t)n * 32 + 16 * hf];
      v16bf bfr;
#pragma unroll
      for (int i = 0; i < 8; ++i) set_pair(bfr, i, bp[i]);
#pragma unroll
      for (int s = 0; s < 4; ++s)
        acc[s][c] = __builtin_amdgcn_wmma_f32_16x16x32_bf16(
            false, a[s], false, bfr, (short)0, acc[s][c], false, false);
    }
    wait_async();
    __syncthreads();
    cur ^= 1;
  }

#pragma unroll
  for (int c = 0; c < 3; ++c) {
    int e = cg * 48 + c * 16 + l16;
#pragma unroll
    for (int s = 0; s < 4; ++s) {
#pragma unroll
      for (int r = 0; r < 8; ++r) {
        int srow = srow0 + s * 16 + r + 8 * hf;
        unsigned short val = f2bf_u16(acc[s][c][r]);
        if (e < 64)
          Q[((size_t)bh * S_ + srow) * HD_ + e] = val;
        else if (e < 128)
          K[((size_t)bh * S_ + srow) * HD_ + (e - 64)] = val;
        else
          Vt[((size_t)bh * HD_ + (e - 128)) * S_ + srow] = val;
      }
    }
  }
}

// ---------------------------------------------------------------------------
// Flash attention. Block = (b, h, 256-row q-block); wave owns 32 q-rows
// (2 row-groups of 16), so every K/V LDS fragment feeds 2 WMMAs.
__global__ __launch_bounds__(256) void attn_kernel(
    const unsigned short* __restrict__ Q,
    const unsigned short* __restrict__ K,
    const unsigned short* __restrict__ Vt,
    unsigned short* __restrict__ Zb)  // [B][S][D] bf16
{
  __shared__ unsigned short kbuf[2][32 * 64];   // [key][e]   4KB x2
  __shared__ unsigned short vbuf[2][64 * 32];   // [e][key]   4KB x2
  __shared__ unsigned short plds[8][32 * 32];   // per-wave P (32x32) 16KB

  const int tid = threadIdx.x;
  const int lane = tid & 31, wv = tid >> 5;
  const int hf = lane >> 4, l16 = lane & 15;
  const int sTiles = S_ / 256;
  const int bh = blockIdx.x / sTiles, st = blockIdx.x % sTiles;
  const int b = bh / H_, h = bh % H_;
  const int q0 = st * 256 + wv * 32;

  const unsigned short* Qb = Q + ((size_t)bh * S_ + q0) * HD_;
  const unsigned short* Kb = K + (size_t)bh * S_ * HD_;
  const unsigned short* Vb = Vt + (size_t)bh * HD_ * S_;

  auto stage_kv = [&](int kb, unsigned short* kdst, unsigned short* vdst) {
    async_load_b128(kdst + (size_t)tid * 8,
                    Kb + ((size_t)(kb + (tid >> 3))) * HD_ + (tid & 7) * 8);
    async_load_b128(vdst + (size_t)tid * 8,
                    Vb + ((size_t)(tid >> 2)) * S_ + kb + (tid & 3) * 8);
  };

  // Q A-fragments: 2 row-groups x 2 e-steps, resident for the whole loop.
  v16bf aq[2][2];
#pragma unroll
  for (int g = 0; g < 2; ++g)
#pragma unroll
    for (int f = 0; f < 2; ++f) {
      const unsigned* qp =
          (const unsigned*)(Qb + (size_t)(g * 16 + l16) * HD_ + f * 32);
#pragma unroll
      for (int i = 0; i < 8; ++i) set_pair(aq[g][f], i, qp[a_k(i, hf) >> 1]);
    }

  float mrow[2][8], lrow[2][8];
#pragma unroll
  for (int g = 0; g < 2; ++g)
#pragma unroll
    for (int r = 0; r < 8; ++r) { mrow[g][r] = -1e30f; lrow[g][r] = 0.f; }
  v8f zac[2][4] = {};
  const float scale = 1.0f / HD_;

  stage_kv(0, kbuf[0], vbuf[0]);
  wait_async();
  __syncthreads();
  int cur = 0;

  for (int kb = 0; kb < S_; kb += 32) {
    if (kb + 32 < S_) stage_kv(kb + 32, kbuf[cur ^ 1], vbuf[cur ^ 1]);

    // ---- scores: each K fragment feeds both row-groups
    v8f sc[2][2] = {};
#pragma unroll
    for (int t = 0; t < 2; ++t) {
#pragma unroll
      for (int f = 0; f < 2; ++f) {
        const unsigned* kp = (const unsigned*)
            &kbuf[cur][(size_t)(t * 16 + l16) * 64 + f * 32 + 16 * hf];
        v16bf bk;
#pragma unroll
        for (int i = 0; i < 8; ++i) set_pair(bk, i, kp[i]);
        sc[0][t] = __builtin_amdgcn_wmma_f32_16x16x32_bf16(
            false, aq[0][f], false, bk, (short)0, sc[0][t], false, false);
        sc[1][t] = __builtin_amdgcn_wmma_f32_16x16x32_bf16(
            false, aq[1][f], false, bk, (short)0, sc[1][t], false, false);
      }
    }
    // ---- online softmax per row-group
#pragma unroll
    for (int g = 0; g < 2; ++g) {
#pragma unroll
      for (int r = 0; r < 8; ++r) {
        float s0 = sc[g][0][r] * scale, s1 = sc[g][1][r] * scale;
        float m = fmaxf(s0, s1);
        m = fmaxf(m, __shfl_xor(m, 1));
        m = fmaxf(m, __shfl_xor(m, 2));
        m = fmaxf(m, __shfl_xor(m, 4));
        m = fmaxf(m, __shfl_xor(m, 8));
        float mnew = fmaxf(mrow[g][r], m);
        float alpha = __expf(mrow[g][r] - mnew);
        float p0 = __expf(s0 - mnew);
        float p1 = __expf(s1 - mnew);
        sc[g][0][r] = p0;
        sc[g][1][r] = p1;
        float rs = p0 + p1;
        rs += __shfl_xor(rs, 1);
        rs += __shfl_xor(rs, 2);
        rs += __shfl_xor(rs, 4);
        rs += __shfl_xor(rs, 8);
        lrow[g][r] = lrow[g][r] * alpha + rs;
        mrow[g][r] = mnew;
#pragma unroll
        for (int t2 = 0; t2 < 4; ++t2) zac[g][t2][r] *= alpha;
      }
      // P: C layout -> LDS (rows g*16..g*16+15)
#pragma unroll
      for (int t = 0; t < 2; ++t)
#pragma unroll
        for (int r = 0; r < 8; ++r)
          plds[wv][(g * 16 + r + 8 * hf) * 32 + t * 16 + l16] =
              f2bf_u16(sc[g][t][r]);
    }
    __syncthreads();
    v16bf ap[2];
#pragma unroll
    for (int g = 0; g < 2; ++g) {
      const unsigned* pp = (const unsigned*)&plds[wv][(g * 16 + l16) * 32];
#pragma unroll
      for (int i = 0; i < 8; ++i) set_pair(ap[g], i, pp[a_k(i, hf) >> 1]);
    }
    __syncthreads();

    // ---- Z += P * V: each V fragment feeds both row-groups
#pragma unroll
    for (int t2 = 0; t2 < 4; ++t2) {
      const unsigned* vp = (const unsigned*)
          &vbuf[cur][(size_t)(t2 * 16 + l16) * 32 + 16 * hf];
      v16bf bv;
#pragma unroll
      for (int i = 0; i < 8; ++i) set_pair(bv, i, vp[i]);
      zac[0][t2] = __builtin_amdgcn_wmma_f32_16x16x32_bf16(
          false, ap[0], false, bv, (short)0, zac[0][t2], false, false);
      zac[1][t2] = __builtin_amdgcn_wmma_f32_16x16x32_bf16(
          false, ap[1], false, bv, (short)0, zac[1][t2], false, false);
    }
    wait_async();
    __syncthreads();
    cur ^= 1;
  }

  // ---- normalize and write Z into [B][S][D] bf16
#pragma unroll
  for (int g = 0; g < 2; ++g)
#pragma unroll
    for (int t2 = 0; t2 < 4; ++t2) {
      int e = t2 * 16 + l16;
#pragma unroll
      for (int r = 0; r < 8; ++r) {
        int srow = q0 + g * 16 + r + 8 * hf;
        float z = zac[g][t2][r] / lrow[g][r];
        Zb[((size_t)b * S_ + srow) * D_ + h * HD_ + e] = f2bf_u16(z);
      }
    }
}

// ---------------------------------------------------------------------------
// Out projection: [8192 x 1024] bf16 x [1024 x 1024] bf16 -> f32.
// Block = 128 rows x 64 cols; wave = 64 rows x 16 cols, so the single LDS
// B-fragment per k-step is reused by 4 WMMAs.
__global__ __launch_bounds__(256) void out_proj_kernel(
    const unsigned short* __restrict__ Zb,      // [8192][1024] bf16
    const unsigned short* __restrict__ wout_t,  // [n][k] bf16
    float* __restrict__ out)                    // [8192][1024] f32
{
  __shared__ unsigned short obuf[2][64 * 32];  // [n][k] panel, 4KB x2

  const int tid = threadIdx.x;
  const int lane = tid & 31, wv = tid >> 5;
  const int hf = lane >> 4, l16 = lane & 15;
  const int rg = wv >> 2, cg = wv & 3;
  const int colBlocks = D_ / 64;
  const int rb = blockIdx.x / colBlocks, cb = blockIdx.x % colBlocks;
  const int row0 = rb * 128 + rg * 64;
  const unsigned short* zbase = Zb + (size_t)row0 * D_;

  auto stage_panel = [&](int k0, unsigned short* dst) {
    async_load_b128(dst + (size_t)tid * 8,
                    wout_t + (size_t)(cb * 64 + (tid >> 2)) * D_ + k0 +
                        (tid & 3) * 8);
  };

  v8f acc[4] = {};
  stage_panel(0, obuf[0]);
  wait_async();
  __syncthreads();
  int cur = 0;

  for (int k0 = 0; k0 < D_; k0 += 32) {
    if (k0 + 32 < D_) stage_panel(k0 + 32, obuf[cur ^ 1]);

    v16bf a[4];
#pragma unroll
    for (int s = 0; s < 4; ++s) {
      const unsigned* ap =
          (const unsigned*)(zbase + (size_t)(s * 16 + l16) * D_ + k0);
#pragma unroll
      for (int i = 0; i < 8; ++i) set_pair(a[s], i, ap[a_k(i, hf) >> 1]);
    }
    const unsigned* bp =
        (const unsigned*)&obuf[cur][(size_t)(cg * 16 + l16) * 32 + 16 * hf];
    v16bf bb;
#pragma unroll
    for (int i = 0; i < 8; ++i) set_pair(bb, i, bp[i]);
#pragma unroll
    for (int s = 0; s < 4; ++s)
      acc[s] = __builtin_amdgcn_wmma_f32_16x16x32_bf16(
          false, a[s], false, bb, (short)0, acc[s], false, false);
    wait_async();
    __syncthreads();
    cur ^= 1;
  }

#pragma unroll
  for (int s = 0; s < 4; ++s) {
    int col = cb * 64 + cg * 16 + l16;
#pragma unroll
    for (int r = 0; r < 8; ++r) {
      int row = row0 + s * 16 + r + 8 * hf;
      out[(size_t)row * D_ + col] = acc[s][r];
    }
  }
}

// ---------------------------------------------------------------------------
extern "C" void kernel_launch(void* const* d_in, const int* in_sizes, int n_in,
                              void* d_out, int out_size, void* d_ws,
                              size_t ws_size, hipStream_t stream) {
  const float* x    = (const float*)d_in[0];
  const float* wqkv = (const float*)d_in[1];
  const float* wout = (const float*)d_in[2];
  float* out = (float*)d_out;

  char* ws = (char*)d_ws;
  size_t off = 0;
  unsigned short* Q  = (unsigned short*)(ws + off); off += (size_t)B_ * H_ * S_ * HD_ * 2;
  unsigned short* K  = (unsigned short*)(ws + off); off += (size_t)B_ * H_ * S_ * HD_ * 2;
  unsigned short* Vt = (unsigned short*)(ws + off); off += (size_t)B_ * H_ * HD_ * S_ * 2;
  unsigned short* Zb = (unsigned short*)(ws + off); off += (size_t)B_ * S_ * D_ * 2;
  unsigned short* Wq = (unsigned short*)(ws + off); off += (size_t)H_ * 192 * D_ * 2;
  unsigned short* Wo = (unsigned short*)(ws + off); off += (size_t)D_ * D_ * 2;
  unsigned short* Xb = (unsigned short*)(ws + off); off += (size_t)B_ * S_ * D_ * 2;
  (void)ws_size; (void)in_sizes; (void)n_in; (void)out_size;

  prep_x_kernel<<<((size_t)B_ * S_ * D_ + 255) / 256, 256, 0, stream>>>(x, Xb);
  prep_wqkv_kernel<<<((size_t)H_ * 192 * D_ + 255) / 256, 256, 0, stream>>>(wqkv, Wq);
  prep_wout_kernel<<<((size_t)D_ * D_ + 255) / 256, 256, 0, stream>>>(wout, Wo);

  const int blocks_qkv = B_ * H_ * (S_ / 128);   // 1024
  qkv_proj_kernel<<<blocks_qkv, 256, 0, stream>>>(Xb, Wq, Q, K, Vt);

  const int blocks_attn = B_ * H_ * (S_ / 256);  // 512
  attn_kernel<<<blocks_attn, 256, 0, stream>>>(Q, K, Vt, Zb);

  const int blocks_out = (B_ * S_ / 128) * (D_ / 64);  // 1024
  out_proj_kernel<<<blocks_out, 256, 0, stream>>>(Zb, Wo, out);
}
